// classifier_8959301779747
// MI455X (gfx1250) — compile-verified, matrix operand order checked by reading, not run on hardware
//
#include <hip/hip_runtime.h>
#include <hip/hip_bf16.h>

#define NN   50000      // nodes
#define NE   800000     // edges
#define DIN  128        // input feature dim = 2H
#define HID  64         // hidden
#define NG   64         // graphs

typedef __attribute__((ext_vector_type(16))) __bf16 v16bf;
typedef __attribute__((ext_vector_type(8)))  float  v8f;

// ================================================================ weight prep:
// f32 -> bf16 AND permute into WMMA B-fragment register layout, so each lane
// loads its 16-element fragment as two contiguous b128s.
// Fragment layout: frag[((ks*CT + ct)*32 + lane)*16 + t]
//   ln = lane&15 (column in tile), kh = lane>>4, a0 = ks*32 + 8*kh
//   t<8 : k = a0 + t ; t>=8 : k = a0 + 16 + (t-8) ; n = ct*16 + ln
__global__ void prep_w1_kernel(const float* __restrict__ W1, __bf16* __restrict__ F) {
    int idx = blockIdx.x * blockDim.x + threadIdx.x;          // 7 * 8192
    if (idx >= 7 * DIN * HID) return;
    int layer = idx >> 13;               // / 8192
    int rem   = idx & 8191;
    int ks    = rem >> 11;               // 4 k-steps
    int rem2  = rem & 2047;
    int ct    = rem2 >> 9;               // 4 col tiles
    int rem3  = rem2 & 511;
    int lane  = rem3 >> 4;
    int t     = rem3 & 15;
    int ln = lane & 15, kh = lane >> 4;
    int a0 = ks * 32 + 8 * kh;
    int k  = (t < 8) ? (a0 + t) : (a0 + 16 + (t - 8));
    int n  = ct * 16 + ln;
    F[idx] = (__bf16)W1[(size_t)layer * DIN * HID + k * HID + n];
}

__global__ void prep_w2_kernel(const float* __restrict__ W2, __bf16* __restrict__ F) {
    int idx = blockIdx.x * blockDim.x + threadIdx.x;          // 6 * 8192
    if (idx >= 6 * HID * DIN) return;
    int layer = idx >> 13;
    int rem   = idx & 8191;
    int ks    = rem >> 12;               // 2 k-steps
    int rem2  = rem & 4095;
    int ct    = rem2 >> 9;               // 8 col tiles
    int rem3  = rem2 & 511;
    int lane  = rem3 >> 4;
    int t     = rem3 & 15;
    int ln = lane & 15, kh = lane >> 4;
    int a0 = ks * 32 + 8 * kh;
    int k  = (t < 8) ? (a0 + t) : (a0 + 16 + (t - 8));
    int n  = ct * 16 + ln;
    F[idx] = (__bf16)W2[(size_t)layer * HID * DIN + k * DIN + n];
}

// ---------------------------------------------------------------- per-graph node counts
__global__ void counts_kernel(const int* __restrict__ batch, int* __restrict__ counts) {
    int n = blockIdx.x * blockDim.x + threadIdx.x;
    if (n < NN) atomicAdd(&counts[batch[n]], 1);
}

// ---------------------------------------------------------------- h = (1+eps[i]) * x
__global__ void hinit_kernel(const float* __restrict__ x, const float* __restrict__ eps,
                             int layer, float* __restrict__ h) {
    int idx = blockIdx.x * blockDim.x + threadIdx.x;     // float4 granularity
    if (idx < NN * (DIN / 4)) {
        float e = 1.0f + eps[layer];
        float4 v = ((const float4*)x)[idx];
        float4 o = {e * v.x, e * v.y, e * v.z, e * v.w};
        ((float4*)h)[idx] = o;
    }
}

// ---------------------------------------------------------------- h[dst] += x[src] (GIN sum agg)
__global__ void scatter_kernel(const float* __restrict__ x, const int* __restrict__ src,
                               const int* __restrict__ dst, float* __restrict__ h) {
    int idx = blockIdx.x * blockDim.x + threadIdx.x;     // 32 threads per edge, 4 ch each
    if (idx < NE * 32) {
        int e  = idx >> 5;
        int cg = (idx & 31) << 2;
        int s = src[e], d = dst[e];
        float4 v = *(const float4*)(x + (size_t)s * DIN + cg);
        float* hp = h + (size_t)d * DIN + cg;
        atomicAdd(hp + 0, v.x); atomicAdd(hp + 1, v.y);
        atomicAdd(hp + 2, v.z); atomicAdd(hp + 3, v.w);
    }
}

// ---------------------------------------------------------------- GEMM1: Z[N,64] = H[N,128] @ W1  (WMMA bf16)
// block = 4 waves; wave w -> 16x16 tile at (blockIdx*16, w*16); fused BN column stats.
__global__ __launch_bounds__(128)
void gemm1_wmma_kernel(const float* __restrict__ A, const __bf16* __restrict__ Bf,
                       float* __restrict__ Z, float* __restrict__ bsum, float* __restrict__ bsq) {
    const int wave = threadIdx.x >> 5;   // = col tile ct
    const int lane = threadIdx.x & 31;
    const int ln = lane & 15;
    const int kh = lane >> 4;
    const int rowBase = blockIdx.x * 16;
    const float* arow = A + (size_t)(rowBase + ln) * DIN;
    v8f acc = {};
    #pragma unroll
    for (int ks = 0; ks < 4; ++ks) {
        const int a0 = ks * 32 + 8 * kh;
        // A: two contiguous 8-float (32B) runs per lane -> b128 loads
        float4 p0 = *(const float4*)(arow + a0);
        float4 p1 = *(const float4*)(arow + a0 + 4);
        float4 p2 = *(const float4*)(arow + a0 + 16);
        float4 p3 = *(const float4*)(arow + a0 + 20);
        v16bf af;
        af[0]  = (__bf16)p0.x; af[1]  = (__bf16)p0.y; af[2]  = (__bf16)p0.z; af[3]  = (__bf16)p0.w;
        af[4]  = (__bf16)p1.x; af[5]  = (__bf16)p1.y; af[6]  = (__bf16)p1.z; af[7]  = (__bf16)p1.w;
        af[8]  = (__bf16)p2.x; af[9]  = (__bf16)p2.y; af[10] = (__bf16)p2.z; af[11] = (__bf16)p2.w;
        af[12] = (__bf16)p3.x; af[13] = (__bf16)p3.y; af[14] = (__bf16)p3.z; af[15] = (__bf16)p3.w;
        // B: pre-shuffled fragment, contiguous 32B per lane
        v16bf bf = *(const v16bf*)(Bf + ((size_t)(ks * 4 + wave) * 32 + lane) * 16);
        acc = __builtin_amdgcn_wmma_f32_16x16x32_bf16(false, af, false, bf,
                                                      (short)0, acc, false, false);
    }
    const int n = wave * 16 + ln;
    float s = 0.f, q = 0.f;
    #pragma unroll
    for (int j = 0; j < 8; ++j) {
        int r = rowBase + j + 8 * kh;
        float v = acc[j];
        if (r < NN) { Z[(size_t)r * HID + n] = v; s += v; q += v * v; }
    }
    atomicAdd(&bsum[n], s);
    atomicAdd(&bsq[n],  q);
}

// ---------------------------------------------------------------- finalize BN: scale/shift from stats
__global__ void fbn_kernel(const float* __restrict__ bsum, const float* __restrict__ bsq,
                           const float* __restrict__ g, const float* __restrict__ b,
                           float* __restrict__ scale, float* __restrict__ shift, int C) {
    int c = threadIdx.x;
    if (c < C) {
        const float invN = 1.0f / (float)NN;
        float mu  = bsum[c] * invN;
        float var = bsq[c] * invN - mu * mu;
        float sc  = g[c] * rsqrtf(var + 1e-5f);
        scale[c] = sc;
        shift[c] = b[c] - mu * sc;
    }
}

// ---------------------------------------------------------------- GEMM2: H2[N,128] = relu(BN1(Z)) @ W2  (WMMA bf16)
// block = 8 waves covering 16x128; BN1 affine + ReLU fused into A-load; fused BN2 stats.
__global__ __launch_bounds__(256)
void gemm2_wmma_kernel(const float* __restrict__ Zin, const float* __restrict__ sc1,
                       const float* __restrict__ sh1, const __bf16* __restrict__ Bf,
                       float* __restrict__ H2, float* __restrict__ bsum, float* __restrict__ bsq) {
    const int wave = threadIdx.x >> 5;   // = col tile ct (0..7)
    const int lane = threadIdx.x & 31;
    const int ln = lane & 15;
    const int kh = lane >> 4;
    const int rowBase = blockIdx.x * 16;
    const float* arow = Zin + (size_t)(rowBase + ln) * HID;
    v8f acc = {};
    #pragma unroll
    for (int ks = 0; ks < 2; ++ks) {
        const int a0 = ks * 32 + 8 * kh;
        float4 p0 = *(const float4*)(arow + a0);
        float4 p1 = *(const float4*)(arow + a0 + 4);
        float4 p2 = *(const float4*)(arow + a0 + 16);
        float4 p3 = *(const float4*)(arow + a0 + 20);
        v16bf af;
        #pragma unroll
        for (int t = 0; t < 4; ++t) {
            float v0 = (&p0.x)[t], v1 = (&p1.x)[t], v2 = (&p2.x)[t], v3 = (&p3.x)[t];
            int k0 = a0 + t, k1 = a0 + 4 + t, k2 = a0 + 16 + t, k3 = a0 + 20 + t;
            af[t]      = (__bf16)fmaxf(sc1[k0] * v0 + sh1[k0], 0.0f);
            af[4 + t]  = (__bf16)fmaxf(sc1[k1] * v1 + sh1[k1], 0.0f);
            af[8 + t]  = (__bf16)fmaxf(sc1[k2] * v2 + sh1[k2], 0.0f);
            af[12 + t] = (__bf16)fmaxf(sc1[k3] * v3 + sh1[k3], 0.0f);
        }
        v16bf bf = *(const v16bf*)(Bf + ((size_t)(ks * 8 + wave) * 32 + lane) * 16);
        acc = __builtin_amdgcn_wmma_f32_16x16x32_bf16(false, af, false, bf,
                                                      (short)0, acc, false, false);
    }
    const int n = wave * 16 + ln;
    float s = 0.f, q = 0.f;
    #pragma unroll
    for (int j = 0; j < 8; ++j) {
        int r = rowBase + j + 8 * kh;
        float v = acc[j];
        if (r < NN) { H2[(size_t)r * DIN + n] = v; s += v; q += v * v; }
    }
    atomicAdd(&bsum[n], s);
    atomicAdd(&bsq[n],  q);
}

// ---------------------------------------------------------------- x = relu(BN2(h2))
__global__ void xnext_kernel(const float* __restrict__ h2, const float* __restrict__ sc,
                             const float* __restrict__ sh, float* __restrict__ xout) {
    int idx = blockIdx.x * blockDim.x + threadIdx.x;     // float4 granularity
    if (idx < NN * (DIN / 4)) {
        int cg = (idx & 31) << 2;
        float4 v = ((const float4*)h2)[idx];
        float4 o;
        o.x = fmaxf(sc[cg + 0] * v.x + sh[cg + 0], 0.0f);
        o.y = fmaxf(sc[cg + 1] * v.y + sh[cg + 1], 0.0f);
        o.z = fmaxf(sc[cg + 2] * v.z + sh[cg + 2], 0.0f);
        o.w = fmaxf(sc[cg + 3] * v.w + sh[cg + 3], 0.0f);
        ((float4*)xout)[idx] = o;
    }
}

// ---------------------------------------------------------------- layer 6 tail: relu(BN1(z)) @ W_last -> relu -> pool-sum
__global__ void last_kernel(const float* __restrict__ Z, const float* __restrict__ sc1,
                            const float* __restrict__ sh1, const float* __restrict__ Wl,
                            const int* __restrict__ batch, float* __restrict__ pool) {
    int nn = blockIdx.x * blockDim.x + threadIdx.x;
    if (nn < NN) {
        const float* z = Z + (size_t)nn * HID;
        float a0 = 0.f, a1 = 0.f;
        #pragma unroll
        for (int k = 0; k < HID; ++k) {
            float a = fmaxf(sc1[k] * z[k] + sh1[k], 0.0f);
            a0 += a * Wl[k * 2 + 0];
            a1 += a * Wl[k * 2 + 1];
        }
        a0 = fmaxf(a0, 0.0f);
        a1 = fmaxf(a1, 0.0f);
        int g = batch[nn];
        atomicAdd(&pool[g * 2 + 0], a0);
        atomicAdd(&pool[g * 2 + 1], a1);
    }
}

// ---------------------------------------------------------------- mean pool divide
__global__ void finalpool_kernel(const float* __restrict__ pool, const int* __restrict__ counts,
                                 float* __restrict__ out) {
    int t = threadIdx.x;
    if (t < NG * 2) {
        float c = (float)counts[t >> 1];
        out[t] = pool[t] / fmaxf(c, 1.0f);
    }
}

extern "C" void kernel_launch(void* const* d_in, const int* in_sizes, int n_in,
                              void* d_out, int out_size, void* d_ws, size_t ws_size,
                              hipStream_t stream) {
    const float* x0    = (const float*)d_in[0];
    const int*   ei    = (const int*)  d_in[1];
    // d_in[2] edge_attribute: unused by reference
    const int*   batch = (const int*)  d_in[3];
    const float* W1    = (const float*)d_in[4];
    const float* bn1g  = (const float*)d_in[5];
    const float* bn1b  = (const float*)d_in[6];
    const float* W2    = (const float*)d_in[7];
    const float* bn2g  = (const float*)d_in[8];
    const float* bn2b  = (const float*)d_in[9];
    const float* Wl    = (const float*)d_in[10];
    const float* eps   = (const float*)d_in[11];
    const int* src = ei;
    const int* dst = ei + NE;

    // ---- workspace layout (256B aligned)
    char* ws = (char*)d_ws;
    size_t off = 0;
    auto walloc = [&](size_t bytes) -> void* {
        void* p = ws + off;
        off = (off + bytes + 255) & ~(size_t)255;
        return p;
    };
    float*   bufA  = (float*)  walloc((size_t)NN * DIN * 4);   // x (layers >= 1)
    float*   bufH  = (float*)  walloc((size_t)NN * DIN * 4);   // h / raw h2
    float*   bufZ  = (float*)  walloc((size_t)NN * HID * 4);   // z (GEMM1 out)
    __bf16*  W1f   = (__bf16*) walloc((size_t)7 * DIN * HID * 2);  // WMMA-layout frags
    __bf16*  W2f   = (__bf16*) walloc((size_t)6 * HID * DIN * 2);
    float*   stats = (float*)  walloc(384 * 4);                // sum1|sq1|sum2|sq2
    float*   aff   = (float*)  walloc(384 * 4);                // sc1|sh1|sc2|sh2
    float*   pool  = (float*)  walloc(NG * 2 * 4);
    int*     cnts  = (int*)    walloc(NG * 4);

    float* sum1 = stats;       float* sq1 = stats + 64;
    float* sum2 = stats + 128; float* sq2 = stats + 256;
    float* sc1  = aff;         float* sh1 = aff + 64;
    float* sc2  = aff + 128;   float* sh2 = aff + 256;

    // ---- one-time prep
    hipMemsetAsync(pool, 0, NG * 2 * 4, stream);
    hipMemsetAsync(cnts, 0, NG * 4, stream);
    prep_w1_kernel<<<(7 * DIN * HID + 255) / 256, 256, 0, stream>>>(W1, W1f);
    prep_w2_kernel<<<(6 * HID * DIN + 255) / 256, 256, 0, stream>>>(W2, W2f);
    counts_kernel<<<(NN + 255) / 256, 256, 0, stream>>>(batch, cnts);

    const int gRows   = NN / 16;              // 3125, exact
    const int gElem4  = (NN * (DIN / 4) + 255) / 256;
    const int gEdge   = (NE * 32 + 255) / 256;

    for (int i = 0; i < 7; ++i) {
        const float* xcur = (i == 0) ? x0 : bufA;
        hipMemsetAsync(stats, 0, 384 * 4, stream);
        hinit_kernel<<<gElem4, 256, 0, stream>>>(xcur, eps, i, bufH);
        scatter_kernel<<<gEdge, 256, 0, stream>>>(xcur, src, dst, bufH);
        gemm1_wmma_kernel<<<gRows, 128, 0, stream>>>(bufH, W1f + (size_t)i * DIN * HID,
                                                     bufZ, sum1, sq1);
        fbn_kernel<<<1, 64, 0, stream>>>(sum1, sq1, bn1g + i * 64, bn1b + i * 64,
                                         sc1, sh1, HID);
        if (i < 6) {
            gemm2_wmma_kernel<<<gRows, 256, 0, stream>>>(bufZ, sc1, sh1,
                                                         W2f + (size_t)i * HID * DIN,
                                                         bufH, sum2, sq2);
            fbn_kernel<<<1, 128, 0, stream>>>(sum2, sq2, bn2g + i * 128, bn2b + i * 128,
                                              sc2, sh2, DIN);
            xnext_kernel<<<gElem4, 256, 0, stream>>>(bufH, sc2, sh2, bufA);
        } else {
            last_kernel<<<(NN + 255) / 256, 256, 0, stream>>>(bufZ, sc1, sh1, Wl, batch, pool);
        }
    }
    finalpool_kernel<<<1, 128, 0, stream>>>(pool, cnts, (float*)d_out);
}
